// ContinuousGaussianHMM_76862734730020
// MI455X (gfx1250) — matrix operand order repaired and checked by reading, not compiled, not run on hardware
//
#include <hip/hip_runtime.h>
#include <math.h>

// ---------------------------------------------------------------------------
// CDNA5 (gfx1250) HMM forward:
//   - transition contraction folded to one bf16 WMMA GEMM  (A' = a_t (x) belief)
//   - observation Gaussian log-prob as f32 WMMA GEMM (cancellation-sensitive)
//   - 16 workgroups (one per 16-row batch tile), persistent over T=200 steps
// ---------------------------------------------------------------------------

typedef __attribute__((ext_vector_type(16))) __bf16  v16bf;
typedef __attribute__((ext_vector_type(8)))  float   v8f;
typedef __attribute__((ext_vector_type(2)))  float   v2f;

#define T_N   200
#define B_N   256
#define S_N   256
#define A_N   16
#define OBS_N 128
#define CTL_N 16

#define EPS_F     1e-6f
#define LOG2PI_F  1.8378770664093453f

// padded LDS row strides (floats) to avoid bank conflicts
#define BS 257   // belief
#define LS 257   // logits
#define XS 258   // xext (even: v2f loads stay 8B aligned)

// dynamic LDS layout (byte offsets, all 32B aligned)
#define SM_APACK   0                       // 128 KB: bf16 A' in WMMA-A packed layout
#define SM_BELIEF  131072                  // 16*257*4 = 16448 (reserve 16480)
#define SM_LOGITS  (SM_BELIEF + 16480)
#define SM_XEXT    (SM_LOGITS + 16480)     // 16*258*4 = 16512
#define SM_ACTP    (SM_XEXT + 16512)       // act_prior [256][16] f32
#define SM_U       (SM_ACTP + 16384)       // u_t tile [16][16]
#define SM_AP      (SM_U + 1024)           // action prior logits [16][16]
#define SM_AT      (SM_AP + 1024)          // a_t [16][16]
#define SM_CIV     (SM_AT + 1024)          // exp(-ctl_lv) [16][16]
#define SM_CMU     (SM_CIV + 1024)         // ctl_mu [16][16]
#define SM_CC      (SM_CMU + 1024)         // ctl const [16]
#define SM_COBS    (SM_CC + 64)            // obs const [256]
#define SM_LOGPI0  (SM_COBS + 1024)        // log_softmax(pi0) [256]
#define SMEM_BYTES (SM_LOGPI0 + 1024)      // = 204160 B  (< 320 KB WGP LDS)

// workspace layout (bytes)
#define WS_WTRANS  0                       // bf16 [4096 K x 256 N] packed tiles: 2 MB
#define WS_BOBS    2097152                 // f32  [256 K x 256 N] packed tiles: 256 KB
#define WS_COBS    (WS_BOBS + 262144)
#define WS_LOGPI0  (WS_COBS + 1024)
#define WS_CIV     (WS_LOGPI0 + 1024)
#define WS_CMU     (WS_CIV + 1024)
#define WS_CC      (WS_CMU + 1024)

__device__ __forceinline__ unsigned bf16_rne(float f) {
  unsigned uu = __float_as_uint(f);
  return (uu + 0x7FFFu + ((uu >> 16) & 1u)) >> 16;
}

// ---------------------------------------------------------------------------
// prep: softmax(trans_logits) -> bf16 packed WMMA-B tiles.
// B' row index Kg = k*S + i ; column j. Tile (nt,kt): lane = half*16+nl,
// n = nt*16+nl, element e -> K = kt*32 + half*16 + e.
// ---------------------------------------------------------------------------
__global__ void prep_trans_kernel(const float* __restrict__ tl, __bf16* __restrict__ wtrans) {
  const int row = blockIdx.x;          // Kg in [0,4096)
  const int lane = threadIdx.x;        // one wave32 per row
  const float* src = tl + (size_t)row * S_N;
  float v[8];
  float mx = -3.4e38f;
  #pragma unroll
  for (int q = 0; q < 8; ++q) { v[q] = src[lane + 32 * q]; mx = fmaxf(mx, v[q]); }
  #pragma unroll
  for (int m = 16; m >= 1; m >>= 1) mx = fmaxf(mx, __shfl_xor(mx, m, 32));
  float s = 0.f;
  #pragma unroll
  for (int q = 0; q < 8; ++q) { v[q] = expf(v[q] - mx); s += v[q]; }
  #pragma unroll
  for (int m = 16; m >= 1; m >>= 1) s += __shfl_xor(s, m, 32);
  const float inv = 1.f / s;
  const int kt = row >> 5, r = row & 31, half = r >> 4, e = r & 15;
  #pragma unroll
  for (int q = 0; q < 8; ++q) {
    int j = lane + 32 * q;
    int nt = j >> 4, nl = j & 15;
    wtrans[((size_t)(nt * 128 + kt) * 32 + half * 16 + nl) * 16 + e] = (__bf16)(v[q] * inv);
  }
}

// ---------------------------------------------------------------------------
// prep: obs weights. Bobs[k,s]: k<128 -> iv ; k>=128 -> -2*mu*iv (pairs with x).
// cobs[s] = sum(mu^2*iv + lv) + 128*log(2pi).  f32 WMMA-B packed layout.
// ---------------------------------------------------------------------------
__global__ void prep_obs_kernel(const float* __restrict__ mu, const float* __restrict__ lv,
                                float* __restrict__ bobs, float* __restrict__ cobs) {
  __shared__ float red[256];
  const int s = blockIdx.x;
  const int k = threadIdx.x;
  const int d = k & 127;
  const float l  = lv[(size_t)s * OBS_N + d];
  const float m  = mu[(size_t)s * OBS_N + d];
  const float iv = expf(-l);
  const float val = (k < OBS_N) ? iv : (-2.f * m * iv);
  const int nt = s >> 4, nl = s & 15, kt = k >> 2, r = k & 3, half = r >> 1, e = r & 1;
  bobs[((size_t)(nt * 64 + kt) * 32 + half * 16 + nl) * 2 + e] = val;
  red[k] = (k < OBS_N) ? (m * m * iv + l) : 0.f;
  __syncthreads();
  for (int st = 128; st > 0; st >>= 1) { if (k < st) red[k] += red[k + st]; __syncthreads(); }
  if (k == 0) cobs[s] = red[0] + (float)OBS_N * LOG2PI_F;
}

// ---------------------------------------------------------------------------
// prep: log_softmax(pi0), ctl constants
// ---------------------------------------------------------------------------
__global__ void prep_misc_kernel(const float* __restrict__ pi0,
                                 const float* __restrict__ ctl_mu, const float* __restrict__ ctl_lv,
                                 float* __restrict__ logpi0, float* __restrict__ civ,
                                 float* __restrict__ cmu, float* __restrict__ cc) {
  __shared__ float red[256];
  const int t = threadIdx.x;
  const float v = pi0[t];
  red[t] = v; __syncthreads();
  for (int st = 128; st > 0; st >>= 1) { if (t < st) red[t] = fmaxf(red[t], red[t + st]); __syncthreads(); }
  const float mx = red[0];
  __syncthreads();
  red[t] = expf(v - mx); __syncthreads();
  for (int st = 128; st > 0; st >>= 1) { if (t < st) red[t] += red[t + st]; __syncthreads(); }
  const float lse = mx + logf(red[0]);
  logpi0[t] = v - lse;
  civ[t] = expf(-ctl_lv[t]);
  cmu[t] = ctl_mu[t];
  if (t < A_N) {
    float s = 0.f;
    for (int d2 = 0; d2 < CTL_N; ++d2) s += ctl_lv[t * CTL_N + d2];
    cc[t] = s + (float)CTL_N * LOG2PI_F;
  }
}

// ---------------------------------------------------------------------------
// device GEMM fragments (one wave per 16-column N tile; wid = nt)
// ---------------------------------------------------------------------------
__device__ __forceinline__ v8f trans_gemm(const __bf16* Apack, const __bf16* __restrict__ wtrans,
                                          int wid, int lane) {
  v8f acc = {};
  const __bf16* ab = Apack + (size_t)lane * 16;
  const __bf16* wb = wtrans + ((size_t)(wid * 128) * 32 + lane) * 16;
  #pragma unroll 4
  for (int kt = 0; kt < 128; ++kt) {
    v16bf av = *(const v16bf*)(ab + (size_t)kt * 512);
    v16bf bv = *(const v16bf*)(wb + (size_t)kt * 512);
    acc = __builtin_amdgcn_wmma_f32_16x16x32_bf16(false, av, false, bv, (short)0, acc, false, false);
  }
  return acc;
}

__device__ __forceinline__ v8f obs_gemm(const float* xext, const float* __restrict__ bobs,
                                        int wid, int lane) {
  const int half = lane >> 4, nl2 = lane & 15;
  v8f acc = {};
  const float* bo = bobs + ((size_t)(wid * 64) * 32 + lane) * 2;
  const float* xr = xext + nl2 * XS + half * 2;
  #pragma unroll 8
  for (int kt = 0; kt < 64; ++kt) {
    v2f a = *(const v2f*)(xr + kt * 4);
    v2f b = *(const v2f*)(bo + (size_t)kt * 64);
    acc = __builtin_amdgcn_wmma_f32_16x16x4_f32(false, a, false, b, (short)0, acc, false, false);
  }
  return acc;
}

__device__ __forceinline__ void fill_x(float* xext, const float* __restrict__ x,
                                       int t, int rb0, int tid) {
  #pragma unroll
  for (int q = 0; q < 4; ++q) {
    int idx = q * 512 + tid;
    int r = idx >> 7;
    int d = idx & 127;
    const float* src = x + (size_t)t * B_N * OBS_N + (size_t)(rb0 + r) * OBS_N + d;
    float xv = *src;
    xext[r * XS + d]         = xv * xv;   // pairs with iv
    xext[r * XS + OBS_N + d] = xv;        // pairs with -2*mu*iv
    if (t + 1 < T_N) __builtin_prefetch(src + B_N * OBS_N, 0, 1);  // global_prefetch_b8
  }
}

__device__ __forceinline__ void row_softmax_store(const float* logits, float* belief,
                                                  float* __restrict__ alpha_b,
                                                  int t, int rb0, int wid, int lane) {
  const int b = wid;                          // 16 waves -> 16 rows
  float v[8];
  float mx = -3.4e38f;
  #pragma unroll
  for (int q = 0; q < 8; ++q) { v[q] = logits[b * LS + lane + 32 * q]; mx = fmaxf(mx, v[q]); }
  #pragma unroll
  for (int m = 16; m >= 1; m >>= 1) mx = fmaxf(mx, __shfl_xor(mx, m, 32));
  float s = 0.f;
  #pragma unroll
  for (int q = 0; q < 8; ++q) { v[q] = expf(v[q] - mx); s += v[q]; }
  #pragma unroll
  for (int m = 16; m >= 1; m >>= 1) s += __shfl_xor(s, m, 32);
  const float inv = 1.f / s;
  float* br = belief + b * BS;
  float* ob = alpha_b + (size_t)t * B_N * S_N + (size_t)(rb0 + b) * S_N;
  #pragma unroll
  for (int q = 0; q < 8; ++q) { int j = lane + 32 * q; float p = v[q] * inv; br[j] = p; ob[j] = p; }
}

// ---------------------------------------------------------------------------
// main persistent kernel: 16 blocks x 512 threads (16 wave32)
// ---------------------------------------------------------------------------
__global__ __launch_bounds__(512, 1)
void hmm_forward_kernel(const float* __restrict__ x, const float* __restrict__ u,
                        const float* __restrict__ act_prior,
                        const __bf16* __restrict__ wtrans, const float* __restrict__ bobs,
                        const float* __restrict__ cobs, const float* __restrict__ logpi0,
                        const float* __restrict__ civ, const float* __restrict__ cmu,
                        const float* __restrict__ cc,
                        float* __restrict__ alpha_b, float* __restrict__ alpha_a) {
  extern __shared__ float4 smem4[];
  char* smem = (char*)smem4;
  __bf16* Apack  = (__bf16*)(smem + SM_APACK);
  float* belief  = (float*)(smem + SM_BELIEF);
  float* logits  = (float*)(smem + SM_LOGITS);
  float* xext    = (float*)(smem + SM_XEXT);
  float* actp    = (float*)(smem + SM_ACTP);
  float* smu     = (float*)(smem + SM_U);
  float* smap    = (float*)(smem + SM_AP);
  float* smat    = (float*)(smem + SM_AT);
  float* sciv    = (float*)(smem + SM_CIV);
  float* scmu    = (float*)(smem + SM_CMU);
  float* scc     = (float*)(smem + SM_CC);
  float* scobs   = (float*)(smem + SM_COBS);
  float* slogpi0 = (float*)(smem + SM_LOGPI0);

  const int tid  = threadIdx.x;
  const int lane = tid & 31;
  const int wid  = tid >> 5;
  const int half = lane >> 4;
  const int nl   = lane & 15;
  const int rb0  = blockIdx.x * 16;

  for (int i = tid; i < S_N * A_N; i += 512) actp[i] = act_prior[i];
  if (tid < 256) {
    sciv[tid]    = civ[tid];
    scmu[tid]    = cmu[tid];
    scobs[tid]   = cobs[tid];
    slogpi0[tid] = logpi0[tid];
  }
  if (tid < A_N) scc[tid] = cc[tid];

  // ---- t = 0: b0 = softmax(gauss_lp(x0) + log_pi0) ----
  fill_x(xext, x, 0, rb0, tid);
  __syncthreads();
  {
    v8f oq = obs_gemm(xext, bobs, wid, lane);
    const int col = wid * 16 + nl;
    #pragma unroll
    for (int r = 0; r < 8; ++r) {
      int brow = r + half * 8;
      logits[brow * LS + col] = -0.5f * (oq[r] + scobs[col]) + slogpi0[col];
    }
  }
  __syncthreads();
  row_softmax_store(logits, belief, alpha_b, 0, rb0, wid, lane);
  __syncthreads();

  for (int t = 1; t < T_N; ++t) {
    // A1: stage x_t and u_{t-1}; action-prior logits  b @ act_prior
    fill_x(xext, x, t, rb0, tid);
    if (tid < 256) {
      int b = tid >> 4, d = tid & 15;
      smu[tid] = u[(size_t)(t - 1) * B_N * CTL_N + (size_t)(rb0 + b) * CTL_N + d];
      float sdot = 0.f;
      const float* br = belief + b * BS;
      for (int i = 0; i < S_N; ++i) sdot = fmaf(br[i], actp[i * A_N + d], sdot);
      smap[tid] = sdot;
    }
    __syncthreads();
    // A2: control-Gaussian log-likelihood per (b,a)
    if (tid < 256) {
      int b = tid >> 4, a = tid & 15;
      float qv = 0.f;
      #pragma unroll
      for (int d = 0; d < CTL_N; ++d) {
        float diff = smu[b * 16 + d] - scmu[a * 16 + d];
        qv = fmaf(diff * diff, sciv[a * 16 + d], qv);
      }
      smat[tid] = -0.5f * (qv + scc[a]);
    }
    __syncthreads();
    // A3: a_t = softmax(lp_u + log(softmax(ap) + eps)); emit alpha_a
    if (tid < 16) {
      const int b = tid;
      float ap[16], vv[16];
      float m1 = -3.4e38f;
      #pragma unroll
      for (int a = 0; a < 16; ++a) { ap[a] = smap[b * 16 + a]; m1 = fmaxf(m1, ap[a]); }
      float s1 = 0.f;
      #pragma unroll
      for (int a = 0; a < 16; ++a) { ap[a] = expf(ap[a] - m1); s1 += ap[a]; }
      const float inv1 = 1.f / s1;
      float m2 = -3.4e38f;
      #pragma unroll
      for (int a = 0; a < 16; ++a) { vv[a] = smat[b * 16 + a] + logf(ap[a] * inv1 + EPS_F); m2 = fmaxf(m2, vv[a]); }
      float s2 = 0.f;
      #pragma unroll
      for (int a = 0; a < 16; ++a) { vv[a] = expf(vv[a] - m2); s2 += vv[a]; }
      const float inv2 = 1.f / s2;
      float* oa = alpha_a + (size_t)(t - 1) * B_N * A_N + (size_t)(rb0 + b) * A_N;
      #pragma unroll
      for (int a = 0; a < 16; ++a) { float p = vv[a] * inv2; smat[b * 16 + a] = p; oa[a] = p; }
    }
    __syncthreads();
    // A4: A'[b, k*256+i] = a_t[b,k]*belief[b,i] in bf16 WMMA-A packed layout
    {
      unsigned* ap32 = (unsigned*)Apack;
      #pragma unroll 4
      for (int qq = 0; qq < 64; ++qq) {
        int slot = qq * 512 + tid;      // dword slot; 32768 total
        int kt = slot >> 8;
        int rem = slot & 255;
        int lane_s = rem >> 3;
        int ep = rem & 7;
        int row = lane_s & 15;
        int hl  = lane_s >> 4;
        int e0 = ep * 2;                // even element: pair stays in same K-group
        int kg = kt * 32 + hl * 16 + e0;
        int ka = kg >> 8;
        int i0 = kg & 255;
        float aw = smat[row * 16 + ka];
        float v0 = aw * belief[row * BS + i0];
        float v1 = aw * belief[row * BS + i0 + 1];
        ap32[slot] = bf16_rne(v0) | (bf16_rne(v1) << 16);
      }
    }
    __syncthreads();
    // D: WMMA GEMMs (EXEC all-ones here) + combine into logits
    {
      v8f tq = trans_gemm(Apack, wtrans, wid, lane);
      v8f oq = obs_gemm(xext, bobs, wid, lane);
      const int col = wid * 16 + nl;
      const float se = (float)S_N * EPS_F;
      #pragma unroll
      for (int r = 0; r < 8; ++r) {
        int brow = r + half * 8;
        logits[brow * LS + col] = -0.5f * (oq[r] + scobs[col]) + logf(tq[r] + se);
      }
    }
    __syncthreads();
    // E: next belief + emit alpha_b
    row_softmax_store(logits, belief, alpha_b, t, rb0, wid, lane);
    __syncthreads();
  }
}

// ---------------------------------------------------------------------------
extern "C" void kernel_launch(void* const* d_in, const int* in_sizes, int n_in,
                              void* d_out, int out_size, void* d_ws, size_t ws_size,
                              hipStream_t stream) {
  (void)in_sizes; (void)n_in; (void)out_size; (void)ws_size;
  const float* x            = (const float*)d_in[0];
  const float* u            = (const float*)d_in[1];
  const float* obs_mu       = (const float*)d_in[2];
  const float* obs_lv       = (const float*)d_in[3];
  const float* ctl_mu       = (const float*)d_in[4];
  const float* ctl_lv       = (const float*)d_in[5];
  const float* trans_logits = (const float*)d_in[6];
  const float* pi0          = (const float*)d_in[7];
  const float* act_prior    = (const float*)d_in[8];

  char* ws = (char*)d_ws;
  __bf16* wtrans = (__bf16*)(ws + WS_WTRANS);
  float*  bobs   = (float*)(ws + WS_BOBS);
  float*  cobs   = (float*)(ws + WS_COBS);
  float*  logpi0 = (float*)(ws + WS_LOGPI0);
  float*  civ    = (float*)(ws + WS_CIV);
  float*  cmu    = (float*)(ws + WS_CMU);
  float*  cc     = (float*)(ws + WS_CC);

  float* alpha_b = (float*)d_out;
  float* alpha_a = alpha_b + (size_t)T_N * B_N * S_N;

  (void)hipFuncSetAttribute((const void*)hmm_forward_kernel,
                            hipFuncAttributeMaxDynamicSharedMemorySize, SMEM_BYTES);

  prep_trans_kernel<<<A_N * S_N, 32, 0, stream>>>(trans_logits, wtrans);
  prep_obs_kernel<<<S_N, 256, 0, stream>>>(obs_mu, obs_lv, bobs, cobs);
  prep_misc_kernel<<<1, 256, 0, stream>>>(pi0, ctl_mu, ctl_lv, logpi0, civ, cmu, cc);
  hmm_forward_kernel<<<B_N / 16, 512, SMEM_BYTES, stream>>>(
      x, u, act_prior, wtrans, bobs, cobs, logpi0, civ, cmu, cc, alpha_b, alpha_a);
}
